// DrBC_73126113182128
// MI455X (gfx1250) — matrix operand order repaired and checked by reading, not compiled, not run on hardware
//
#include <hip/hip_runtime.h>
#include <hip/hip_bf16.h>

#define Nn 100000
#define Ee 600000
#define Hh 128
#define GDEPTH 6
#define TILES 2            // 16-node M-tiles per block (32 nodes/block)
#define ASTR 136           // padded LDS bf16 row stride (halves): 68 dwords -> conflict-free

typedef __attribute__((ext_vector_type(16))) __bf16 v16bf;
typedef __attribute__((ext_vector_type(8)))  float  v8f;
typedef __attribute__((ext_vector_type(8)))  unsigned int v8u;

// exact pointee type expected by the async-to-LDS builtins: int __vector(4)
typedef int v4i_vs __attribute__((vector_size(16)));
typedef __attribute__((address_space(1))) v4i_vs* g4ptr;   // global int4*
typedef __attribute__((address_space(3))) v4i_vs* s4ptr;   // LDS int4*

#if __has_builtin(__builtin_amdgcn_global_load_async_to_lds_b128)
#define HAVE_ASYNC 1
#else
#define HAVE_ASYNC 0
#endif

// ---- helpers -------------------------------------------------------------

__device__ __forceinline__ unsigned pk2bf(float a, float b) {
#if __has_builtin(__builtin_amdgcn_cvt_pk_bf16_f32)
  auto r = __builtin_amdgcn_cvt_pk_bf16_f32(a, b);   // lo = a, hi = b
  return __builtin_bit_cast(unsigned, r);
#else
  unsigned ua = __float_as_uint(a);
  unsigned ub = __float_as_uint(b);
  ua += 0x7FFFu + ((ua >> 16) & 1u);
  ub += 0x7FFFu + ((ub >> 16) & 1u);
  return (ua >> 16) | (ub & 0xFFFF0000u);
#endif
}

__global__ void fill_kernel(float* __restrict__ p, float v, size_t n) {
  size_t t = (size_t)blockIdx.x * 256 + threadIdx.x;
  if (t < n) p[t] = v;
}

__global__ void cvt_bf16_kernel(const float* __restrict__ w,
                                unsigned short* __restrict__ o, int n) {
  int t = blockIdx.x * 256 + threadIdx.x;
  if (t < n) {
    unsigned u = __float_as_uint(w[t]);
    u += 0x7FFFu + ((u >> 16) & 1u);
    o[t] = (unsigned short)(u >> 16);
  }
}

__global__ void deg_kernel(const int* __restrict__ ei, float* __restrict__ deg, int n2e) {
  int t = blockIdx.x * 256 + threadIdx.x;
  if (t < n2e) atomicAdd(&deg[ei[t]], 1.0f);
}

__global__ void invsqrt_kernel(float* __restrict__ d, int n) {
  int t = blockIdx.x * 256 + threadIdx.x;
  if (t < n) d[t] = rsqrtf(d[t] + 1.0f);   // deg+1 >= 1 -> never inf
}

__global__ void norm_kernel(const int* __restrict__ row, const int* __restrict__ col,
                            const float* __restrict__ dis, float* __restrict__ nrm, int E) {
  int t = blockIdx.x * 256 + threadIdx.x;
  if (t < E) nrm[t] = dis[row[t]] * dis[col[t]];
}

// h = leaky_relu(X @ W0^T + b0); h_max = -inf
__global__ void init_kernel(const float* __restrict__ X, const float* __restrict__ W0,
                            const float* __restrict__ b0, float* __restrict__ h,
                            float* __restrict__ hmax) {
  size_t t = (size_t)blockIdx.x * 256 + threadIdx.x;
  if (t >= (size_t)Nn * Hh) return;
  int f = (int)(t & (Hh - 1));
  size_t node = t >> 7;
  float a = b0[f] + X[node * 3 + 0] * W0[f * 3 + 0]
                  + X[node * 3 + 1] * W0[f * 3 + 1]
                  + X[node * 3 + 2] * W0[f * 3 + 2];
  h[t] = a > 0.f ? a : 0.01f * a;
  hmax[t] = -INFINITY;
}

// h_agg[col[e]] += norm[e] * h[row[e]]  (32 threads/edge, float4 per thread)
__global__ void agg_kernel(const int* __restrict__ row, const int* __restrict__ col,
                           const float* __restrict__ nrm, const float* __restrict__ h,
                           float* __restrict__ hagg) {
  unsigned t = blockIdx.x * 256u + threadIdx.x;
  int e = (int)(t >> 5);
  if (e >= Ee) return;
  int c = (int)(t & 31u) << 2;
  int s = row[e], d = col[e];
  float w = nrm[e];
  const float4 v = *(const float4*)(h + (size_t)s * Hh + c);
  float* dst = hagg + (size_t)d * Hh + c;
  atomicAdd(dst + 0, w * v.x);
  atomicAdd(dst + 1, w * v.y);
  atomicAdd(dst + 2, w * v.z);
  atomicAdd(dst + 3, w * v.w);
}

// ---- WMMA GRU ------------------------------------------------------------
// One block = TILES 16-node tiles. 8 waves; wave w owns feature slice f0=16w.
// gx = h_agg @ W_ih^T, gh = h @ W_hh^T via v_wmma_f32_16x16x32_bf16.
// Tiles staged (async if available) to LDS f32, converted once to padded bf16.

__device__ __forceinline__ v16bf fragA_lds(const unsigned short* sbf,
                                           int mA, int loffA, int kb) {
  // CDNA5 16-bit A layout: lanes 0-15 M=lane K 0-7/16-23; lanes 16-31 K 8-15/24-31
  const uint4 x0 = *(const uint4*)(sbf + mA * ASTR + kb + loffA);
  const uint4 x1 = *(const uint4*)(sbf + mA * ASTR + kb + 16 + loffA);
  v8u u;
  u[0] = x0.x; u[1] = x0.y; u[2] = x0.z; u[3] = x0.w;
  u[4] = x1.x; u[5] = x1.y; u[6] = x1.z; u[7] = x1.w;
  return __builtin_bit_cast(v16bf, u);
}

__device__ __forceinline__ v16bf fragB_bf16(const unsigned short* w, int j,
                                            int kb, int koffB) {
  // CDNA5 16-bit B layout: lane -> N, lanes 0-15 K 0-7/16-23, lanes 16-31 K 8-15/24-31
  const uint4 x0 = *(const uint4*)(w + (size_t)j * Hh + kb + koffB);
  const uint4 x1 = *(const uint4*)(w + (size_t)j * Hh + kb + koffB + 16);
  v8u u;
  u[0] = x0.x; u[1] = x0.y; u[2] = x0.z; u[3] = x0.w;
  u[4] = x1.x; u[5] = x1.y; u[6] = x1.z; u[7] = x1.w;
  return __builtin_bit_cast(v16bf, u);
}

__global__ __launch_bounds__(256)
void gru_kernel(const float* __restrict__ hagg, float* __restrict__ h,
                float* __restrict__ hmax,
                const unsigned short* __restrict__ wih,
                const unsigned short* __restrict__ whh,
                const float* __restrict__ b_ih, const float* __restrict__ b_hh) {
  __shared__ float sAggF[TILES * 16 * Hh];            // 16 KB
  __shared__ float sHF[TILES * 16 * Hh];              // 16 KB (also h_old source)
  __shared__ unsigned short sAggB[TILES * 16 * ASTR]; // padded bf16
  __shared__ unsigned short sHB[TILES * 16 * ASTR];

  const int tid = threadIdx.x;
  const size_t base = (size_t)blockIdx.x * (TILES * 16 * Hh);

  // ---- stage f32 tiles into LDS ----
#if HAVE_ASYNC
  {
    const float* gA = hagg + base;
    const float* gH = h + base;
#pragma unroll
    for (int q = 0; q < TILES * 16 * Hh / (4 * 256); ++q) {
      int i = (tid + q * 256) * 4;   // float index, 16B granules
      __builtin_amdgcn_global_load_async_to_lds_b128(
          (g4ptr)(gA + i), (s4ptr)(sAggF + i), 0, 0);
      __builtin_amdgcn_global_load_async_to_lds_b128(
          (g4ptr)(gH + i), (s4ptr)(sHF + i), 0, 0);
    }
#if __has_builtin(__builtin_amdgcn_s_wait_asynccnt)
    __builtin_amdgcn_s_wait_asynccnt(0);
#else
    asm volatile("s_wait_asynccnt 0" ::: "memory");
#endif
  }
#else
  {
    const float4* gA = (const float4*)(hagg + base);
    const float4* gH = (const float4*)(h + base);
    float4* sA4 = (float4*)sAggF;
    float4* sH4 = (float4*)sHF;
    for (int i = tid; i < TILES * 16 * Hh / 4; i += 256) {
      sA4[i] = gA[i];
      sH4[i] = gH[i];
    }
  }
#endif
  __syncthreads();

  // ---- convert once to padded bf16 (shared across all 8 waves) ----
  {
    const int s0 = tid * 16;            // 16 consecutive floats, within one row
    const int row = s0 >> 7;
    const int cp = s0 & 127;
    unsigned* dA = (unsigned*)(sAggB + row * ASTR + cp);
    unsigned* dH = (unsigned*)(sHB + row * ASTR + cp);
    const float* pA = sAggF + s0;
    const float* pH = sHF + s0;
#pragma unroll
    for (int q = 0; q < 8; ++q) {
      dA[q] = pk2bf(pA[2 * q], pA[2 * q + 1]);
      dH[q] = pk2bf(pH[2 * q], pH[2 * q + 1]);
    }
  }
  __syncthreads();

  const int lane  = tid & 31;
  const int wv    = tid >> 5;
  const int f0    = wv * 16;
  const int mA    = lane & 15;             // A: lane -> row M
  const int loffA = (lane >> 4) << 3;      // A: K sub-offset 0 / 8
  const int nB    = lane & 15;             // B: lane -> col N
  const int koffB = (lane >> 4) << 3;      // B: K sub-offset 0 / 8

  v8f accX[TILES][3] = {};   // xr, xz, xn slices per tile
  v8f accH[TILES][3] = {};   // hr, hz, hn slices per tile

#pragma unroll
  for (int kk = 0; kk < 4; ++kk) {
    const int kb = kk * 32;
    v16bf aA[TILES], aHH[TILES];
#pragma unroll
    for (int t = 0; t < TILES; ++t) {
      aA[t]  = fragA_lds(sAggB + t * 16 * ASTR, mA, loffA, kb);
      aHH[t] = fragA_lds(sHB + t * 16 * ASTR, mA, loffA, kb);
    }
#pragma unroll
    for (int g = 0; g < 3; ++g) {
      const int j = f0 + g * Hh + nB;
      v16bf bx = fragB_bf16(wih, j, kb, koffB);
      v16bf bh = fragB_bf16(whh, j, kb, koffB);
#pragma unroll
      for (int t = 0; t < TILES; ++t) {
        accX[t][g] = __builtin_amdgcn_wmma_f32_16x16x32_bf16(
            false, aA[t], false, bx, (short)0, accX[t][g], false, false);
        accH[t][g] = __builtin_amdgcn_wmma_f32_16x16x32_bf16(
            false, aHH[t], false, bh, (short)0, accH[t][g], false, false);
      }
    }
  }

  // Gate math on the C/D layout: lane&15 = N (feature), VGPR r -> M = r + 8*(lane>>4)
  const int f = f0 + nB;
  const float bir = b_ih[f], biz = b_ih[f + Hh], bin = b_ih[f + 2 * Hh];
  const float bhr = b_hh[f], bhz = b_hh[f + Hh], bhn = b_hh[f + 2 * Hh];
  const int mgrp = (lane >> 4) << 3;
#pragma unroll
  for (int t = 0; t < TILES; ++t) {
#pragma unroll
    for (int r = 0; r < 8; ++r) {
      const int mm = r + mgrp;
      float xr = accX[t][0][r] + bir, xz = accX[t][1][r] + biz, xn = accX[t][2][r] + bin;
      float hr = accH[t][0][r] + bhr, hz = accH[t][1][r] + bhz, hn = accH[t][2][r] + bhn;
      float rg = 1.0f / (1.0f + __expf(-(xr + hr)));
      float zg = 1.0f / (1.0f + __expf(-(xz + hz)));
      float ng = tanhf(xn + rg * hn);
      float hold = sHF[t * 16 * Hh + mm * Hh + f];
      float hnew = (1.0f - zg) * ng + zg * hold;
      size_t gi = base + (size_t)(t * 16 + mm) * Hh + f;
      h[gi] = hnew;
      hmax[gi] = fmaxf(hmax[gi], hnew);
    }
  }
}

// ---- final MLP: out = (leakyrelu(hmax @ Wm1^T + bm1)) @ Wm2^T + bm2 ------
__global__ void mlp_kernel(const float* __restrict__ hmax, const float* __restrict__ Wm1,
                           const float* __restrict__ bm1, const float* __restrict__ Wm2,
                           const float* __restrict__ bm2, float* __restrict__ out) {
  __shared__ float sW[64 * Hh];
  __shared__ float sHid[256];
  for (int i = threadIdx.x; i < 64 * Hh; i += 256) sW[i] = Wm1[i];
  __syncthreads();
  const int node = blockIdx.x * 4 + (threadIdx.x >> 6);
  const int j = threadIdx.x & 63;
  float acc = bm1[j];
  if (node < Nn) {
    const float* hr = hmax + (size_t)node * Hh;
    const float* wr = sW + j * Hh;
#pragma unroll 4
    for (int k = 0; k < Hh; ++k) acc += hr[k] * wr[k];
  }
  float hid = acc > 0.f ? acc : 0.01f * acc;
  sHid[threadIdx.x] = hid * Wm2[j];
  __syncthreads();
  if (j == 0 && node < Nn) {
    float s = bm2[0];
    const float* p = sHid + threadIdx.x;
    for (int k = 0; k < 64; ++k) s += p[k];
    out[node] = s;
  }
}

// ---- launch --------------------------------------------------------------

extern "C" void kernel_launch(void* const* d_in, const int* in_sizes, int n_in,
                              void* d_out, int out_size, void* d_ws, size_t ws_size,
                              hipStream_t stream) {
  const float* X    = (const float*)d_in[0];
  const int*   ei   = (const int*)d_in[1];   // [2, E] flat; row = ei, col = ei + E
  const float* W0   = (const float*)d_in[2];
  const float* b0   = (const float*)d_in[3];
  const float* W_ih = (const float*)d_in[4];
  const float* W_hh = (const float*)d_in[5];
  const float* b_ih = (const float*)d_in[6];
  const float* b_hh = (const float*)d_in[7];
  const float* Wm1  = (const float*)d_in[8];
  const float* bm1  = (const float*)d_in[9];
  const float* Wm2  = (const float*)d_in[10];
  const float* bm2  = (const float*)d_in[11];
  float* out = (float*)d_out;

  float* ws = (float*)d_ws;
  const size_t NH = (size_t)Nn * Hh;
  float* h    = ws;
  float* hagg = ws + NH;
  float* hmax = ws + 2 * NH;
  float* dis  = ws + 3 * NH;                 // degree accumulator, then deg^-1/2
  float* nrm  = ws + 3 * NH + Nn;
  unsigned short* wihb = (unsigned short*)(ws + 3 * NH + Nn + Ee);
  unsigned short* whhb = wihb + 3 * Hh * Hh;

  const int* row = ei;
  const int* col = ei + Ee;

  fill_kernel<<<(Nn + 255) / 256, 256, 0, stream>>>(dis, 0.0f, (size_t)Nn);
  cvt_bf16_kernel<<<(3 * Hh * Hh + 255) / 256, 256, 0, stream>>>(W_ih, wihb, 3 * Hh * Hh);
  cvt_bf16_kernel<<<(3 * Hh * Hh + 255) / 256, 256, 0, stream>>>(W_hh, whhb, 3 * Hh * Hh);
  deg_kernel<<<(2 * Ee + 255) / 256, 256, 0, stream>>>(ei, dis, 2 * Ee);
  invsqrt_kernel<<<(Nn + 255) / 256, 256, 0, stream>>>(dis, Nn);
  norm_kernel<<<(Ee + 255) / 256, 256, 0, stream>>>(row, col, dis, nrm, Ee);
  init_kernel<<<(int)((NH + 255) / 256), 256, 0, stream>>>(X, W0, b0, h, hmax);

  for (int it = 0; it < GDEPTH - 1; ++it) {
    fill_kernel<<<(int)((NH + 255) / 256), 256, 0, stream>>>(hagg, 0.0f, NH);
    agg_kernel<<<(Ee * 32 + 255) / 256, 256, 0, stream>>>(row, col, nrm, h, hagg);
    gru_kernel<<<Nn / (16 * TILES), 256, 0, stream>>>(hagg, h, hmax, wihb, whhb, b_ih, b_hh);
  }

  mlp_kernel<<<(Nn + 3) / 4, 256, 0, stream>>>(hmax, Wm1, bm1, Wm2, bm2, out);
}
